// HiCFL_25786983645193
// MI455X (gfx1250) — matrix operand order repaired and checked by reading, not compiled, not run on hardware
//
#include <hip/hip_runtime.h>
#include <math.h>

#define N_NODES 50000
#define E_EDGES 800000
#define IN_C    512
#define HID     256
#define OUT_CH  2
#define BN_EPS  1e-5f

typedef __bf16 bf16_t;
typedef __attribute__((ext_vector_type(16))) __bf16 v16bf;
typedef __attribute__((ext_vector_type(8)))  __bf16 v8bf;
typedef __attribute__((ext_vector_type(8)))  float  v8f;

__device__ __forceinline__ unsigned short f32_to_bf16_bits(float f) {
  unsigned int u = __float_as_uint(f);
  u += 0x7FFFu + ((u >> 16) & 1u);           // round-to-nearest-even
  return (unsigned short)(u >> 16);
}

// ---------------- utility kernels ----------------
__global__ void zero_f32(float* p, int n) {
  int i = blockIdx.x * blockDim.x + threadIdx.x;
  if (i < n) p[i] = 0.f;
}

__global__ void deg_count(const long long* __restrict__ rows, float* __restrict__ deg, int e) {
  int i = blockIdx.x * blockDim.x + threadIdx.x;
  if (i < e) atomicAdd(&deg[(int)rows[i]], 1.0f);
}

__global__ void dinv_finalize(float* d, int n) {
  int i = blockIdx.x * blockDim.x + threadIdx.x;
  if (i < n) d[i] = rsqrtf(d[i] + 1.0f);     // +1 = self loop; always > 0
}

__global__ void cvt_bf16(const float* __restrict__ src, unsigned short* __restrict__ dst, int n) {
  int i = blockIdx.x * blockDim.x + threadIdx.x;
  if (i < n) dst[i] = f32_to_bf16_bits(src[i]);
}

// dst[n][0:256] = a[n][:], dst[n][256:512] = b[n][:]   (bf16), n over nodes
__global__ void cvt_concat_bf16(const float* __restrict__ a, const float* __restrict__ b,
                                unsigned short* __restrict__ dst, int n) {
  int i = blockIdx.x * blockDim.x + threadIdx.x;
  if (i < n) {
    int node = i >> 9, c = i & 511;
    float v = (c < HID) ? a[(size_t)node * HID + c] : b[(size_t)node * HID + (c - HID)];
    dst[i] = f32_to_bf16_bits(v);
  }
}

// wt[n*K + k] = bf16(W[k*HID + n])  (transpose HIDxK <- KxHID)
__global__ void cvt_wT_bf16(const float* __restrict__ W, unsigned short* __restrict__ wt, int K) {
  int i = blockIdx.x * blockDim.x + threadIdx.x;
  int total = HID * K;
  if (i < total) {
    int n = i / K, k = i - n * K;
    wt[i] = f32_to_bf16_bits(W[(size_t)k * HID + n]);
  }
}

// ---------------- WMMA GEMM: C[M,HID] = A[M,K] @ B[K,HID] (+bias) ----------------
// A row-major bf16 [M,K]; BT = B^T row-major bf16 [HID,K].
// One wave computes a 16(M) x 64(N) strip: 4 accumulators reusing one A fragment.
__global__ void gemm_wmma_bf16(const unsigned short* __restrict__ Abits,
                               const unsigned short* __restrict__ BTbits,
                               const float* __restrict__ bias,
                               float* __restrict__ C, int M, int K) {
  const bf16_t* A  = (const bf16_t*)Abits;
  const bf16_t* BT = (const bf16_t*)BTbits;
  int gt   = blockIdx.x * blockDim.x + threadIdx.x;
  int wave = gt >> 5;
  int lane = gt & 31;
  const int NG = (HID / 16) / 4;           // 4 groups of 4 column tiles
  int mt = wave / NG;
  int ng = wave - mt * NG;
  if (mt * 16 >= M) return;
  int half = lane >> 4;                    // 0/1 : K-half selector
  int l    = lane & 15;

  const bf16_t* ap = A  + (size_t)(mt * 16 + l) * K + half * 16;       // A row m=l
  const bf16_t* bp = BT + (size_t)(ng * 64 + l) * K + half * 16;       // B col n=l (tile j adds 16*K)
  const size_t  K16 = (size_t)16 * K;

  v8f acc0 = {}, acc1 = {}, acc2 = {}, acc3 = {};
  for (int k0 = 0; k0 < K; k0 += 32) {
    v16bf a = *(const v16bf*)(ap + k0);                // K = k0 + half*16 + [0,16)
    v16bf b0 = *(const v16bf*)(bp + k0);
    v16bf b1 = *(const v16bf*)(bp + K16 + k0);
    v16bf b2 = *(const v16bf*)(bp + 2 * K16 + k0);
    v16bf b3 = *(const v16bf*)(bp + 3 * K16 + k0);
    acc0 = __builtin_amdgcn_wmma_f32_16x16x32_bf16(false, a, false, b0, (short)0, acc0, false, false);
    acc1 = __builtin_amdgcn_wmma_f32_16x16x32_bf16(false, a, false, b1, (short)0, acc1, false, false);
    acc2 = __builtin_amdgcn_wmma_f32_16x16x32_bf16(false, a, false, b2, (short)0, acc2, false, false);
    acc3 = __builtin_amdgcn_wmma_f32_16x16x32_bf16(false, a, false, b3, (short)0, acc3, false, false);
  }
  int m0 = mt * 16 + half * 8;
  v8f accs[4] = {acc0, acc1, acc2, acc3};
#pragma unroll
  for (int j = 0; j < 4; ++j) {
    int   col = ng * 64 + j * 16 + l;
    float bv  = bias ? bias[col] : 0.f;
#pragma unroll
    for (int r = 0; r < 8; ++r)
      C[(size_t)(m0 + r) * HID + col] = accs[j][r] + bv;
  }
}

// ---------------- GCN propagate ----------------
// h[n][c] = bias[c] + t[n][c] * dinv[n]^2   (self-loop term)
__global__ void prop_init(const float* __restrict__ t, const float* __restrict__ dinv,
                          const float* __restrict__ bias, float* __restrict__ h, int n) {
  int i = blockIdx.x * blockDim.x + threadIdx.x;
  if (i < n) {
    int node = i >> 8, c = i & (HID - 1);
    float d = dinv[node];
    h[i] = bias[c] + t[i] * d * d;
  }
}

// one block (256 lanes) per edge: h[r][c] += t[s][c] * dinv[r]*dinv[s]
__global__ void prop_edges(const long long* __restrict__ rows, const long long* __restrict__ cols,
                           const float* __restrict__ dinv, const float* __restrict__ t,
                           float* __restrict__ h) {
  int e = blockIdx.x;
  int r = (int)rows[e];
  int s = (int)cols[e];
  float w = dinv[r] * dinv[s];
  int c = threadIdx.x;
  atomicAdd(&h[(size_t)r * HID + c], t[(size_t)s * HID + c] * w);
}

// ---------------- BatchNorm (eval stats over N) + ReLU (+ fused bf16 emit) ----------------
__global__ void bn_stats(const float* __restrict__ x, float* __restrict__ sums) {
  int c = threadIdx.x;                      // 256 cols
  float s = 0.f, s2 = 0.f;
  for (int r = blockIdx.x; r < N_NODES; r += gridDim.x) {
    float v = x[(size_t)r * HID + c];
    s += v; s2 += v * v;
  }
  atomicAdd(&sums[c], s);
  atomicAdd(&sums[HID + c], s2);
}

__global__ void bn_apply_relu(const float* __restrict__ x, const float* __restrict__ sums,
                              const float* __restrict__ g, const float* __restrict__ b,
                              float* __restrict__ y, unsigned short* __restrict__ y16, int n) {
  int i = blockIdx.x * blockDim.x + threadIdx.x;
  if (i < n) {
    int c = i & (HID - 1);
    const float invn = 1.0f / (float)N_NODES;
    float m   = sums[c] * invn;
    float var = sums[HID + c] * invn - m * m;
    float r   = rsqrtf(var + BN_EPS);
    float v   = fmaxf(g[c] * (x[i] - m) * r + b[c], 0.f);
    y[i] = v;
    if (y16) y16[i] = f32_to_bf16_bits(v);
  }
}

// ---------------- 256->2 head + log_softmax ----------------
__global__ void head_logsoftmax(const float* __restrict__ Hin, const float* __restrict__ W,
                                const float* __restrict__ b, float* __restrict__ out) {
  int n = blockIdx.x * blockDim.x + threadIdx.x;
  if (n >= N_NODES) return;
  const float* hr = Hin + (size_t)n * HID;
  float z0 = b[0], z1 = b[1];
#pragma unroll 8
  for (int k = 0; k < HID; ++k) {
    float v = hr[k];
    z0 += v * W[2 * k];
    z1 += v * W[2 * k + 1];
  }
  float m   = fmaxf(z0, z1);
  float lse = m + logf(expf(z0 - m) + expf(z1 - m));
  out[2 * n]     = z0 - lse;
  out[2 * n + 1] = z1 - lse;
}

// ---------------- launcher ----------------
extern "C" void kernel_launch(void* const* d_in, const int* in_sizes, int n_in,
                              void* d_out, int out_size, void* d_ws, size_t ws_size,
                              hipStream_t stream) {
  (void)in_sizes; (void)n_in; (void)out_size; (void)ws_size;
  const float*     x       = (const float*)d_in[0];
  const long long* ei      = (const long long*)d_in[1];
  const float*     conv_w0 = (const float*)d_in[2];
  const float*     conv_ws = (const float*)d_in[3];
  const float*     conv_b  = (const float*)d_in[4];
  const float*     bn_g    = (const float*)d_in[5];
  const float*     bn_b    = (const float*)d_in[6];
  const float*     fcg_w0  = (const float*)d_in[7];
  const float*     fcg_ws  = (const float*)d_in[8];
  const float*     fcg_b   = (const float*)d_in[9];
  const float*     bng_g   = (const float*)d_in[10];
  const float*     bng_b   = (const float*)d_in[11];
  const float*     fcl_w   = (const float*)d_in[12];
  const float*     fcl_b   = (const float*)d_in[13];
  const float*     bnl_g   = (const float*)d_in[14];
  const float*     bnl_b   = (const float*)d_in[15];
  const float*     outg_w  = (const float*)d_in[16];
  const float*     outg_b  = (const float*)d_in[17];
  const float*     outl_w  = (const float*)d_in[18];
  const float*     outl_b  = (const float*)d_in[19];
  float* out = (float*)d_out;

  const long long* rows = ei;
  const long long* cols = ei + E_EDGES;

  // workspace carve-out
  size_t off = 0;
  auto alloc = [&](size_t bytes) -> char* {
    char* p = (char*)d_ws + off;
    off += (bytes + 255) & ~(size_t)255;
    return p;
  };
  float* dinv = (float*)alloc((size_t)N_NODES * 4);
  float* sums = (float*)alloc(2 * HID * 4);
  float* t0   = (float*)alloc((size_t)N_NODES * HID * 4);
  float* t1   = (float*)alloc((size_t)N_NODES * HID * 4);
  float* h    = (float*)alloc((size_t)N_NODES * HID * 4);
  float* hgA  = (float*)alloc((size_t)N_NODES * HID * 4);
  float* hgB  = (float*)alloc((size_t)N_NODES * HID * 4);
  unsigned short* abf = (unsigned short*)alloc((size_t)N_NODES * IN_C * 2);
  unsigned short* wt  = (unsigned short*)alloc((size_t)IN_C * HID * 2);

  dim3 B(256);
  auto nb = [](size_t n) { return dim3((unsigned)((n + 255) / 256)); };

  auto gemm = [&](const unsigned short* A, const unsigned short* BT, const float* bias,
                  float* C, int K) {
    int waves = (N_NODES / 16) * ((HID / 16) / 4);   // 12500 (16x64 strip per wave)
    gemm_wmma_bf16<<<nb((size_t)waves * 32), B, 0, stream>>>(A, BT, bias, C, N_NODES, K);
  };
  auto bn = [&](const float* src, const float* g, const float* bb, float* dst,
                unsigned short* dst16) {
    zero_f32<<<nb(2 * HID), B, 0, stream>>>(sums, 2 * HID);
    bn_stats<<<dim3(512), B, 0, stream>>>(src, sums);
    bn_apply_relu<<<nb((size_t)N_NODES * HID), B, 0, stream>>>(src, sums, g, bb, dst, dst16,
                                                               N_NODES * HID);
  };

  // ---- GCN normalization: dinv = rsqrt(deg + 1) ----
  zero_f32<<<nb(N_NODES), B, 0, stream>>>(dinv, N_NODES);
  deg_count<<<nb(E_EDGES), B, 0, stream>>>(rows, dinv, E_EDGES);
  dinv_finalize<<<nb(N_NODES), B, 0, stream>>>(dinv, N_NODES);

  // ---- GCN conv layers (abf always holds bf16 of the current GEMM A input) ----
  cvt_bf16<<<nb((size_t)N_NODES * IN_C), B, 0, stream>>>(x, abf, N_NODES * IN_C);
  for (int i = 0; i < 3; ++i) {
    int K = (i == 0) ? IN_C : HID;
    const float* W = (i == 0) ? conv_w0 : conv_ws + (size_t)(i - 1) * HID * HID;
    cvt_wT_bf16<<<nb((size_t)HID * K), B, 0, stream>>>(W, wt, K);
    gemm(abf, wt, nullptr, t0, K);
    prop_init<<<nb((size_t)N_NODES * HID), B, 0, stream>>>(t0, dinv, conv_b + i * HID, h,
                                                           N_NODES * HID);
    prop_edges<<<dim3(E_EDGES), B, 0, stream>>>(rows, cols, dinv, t0, h);
    bn(h, bn_g + i * HID, bn_b + i * HID, h, abf);   // abf <- bf16(h) for next GEMM
  }

  // ---- local head helper: assumes abf == bf16(hg_i) on entry ----
  auto local_head = [&](int i, float* outp) {
    cvt_wT_bf16<<<nb((size_t)HID * HID), B, 0, stream>>>(fcl_w + (size_t)i * HID * HID, wt, HID);
    gemm(abf, wt, fcl_b + i * HID, t0, HID);
    bn(t0, bnl_g + i * HID, bnl_b + i * HID, t1, nullptr);
    head_logsoftmax<<<nb(N_NODES), B, 0, stream>>>(t1, outl_w + (size_t)i * HID * OUT_CH,
                                                   outl_b + i * OUT_CH, outp);
  };

  // ---- hierarchical global path ----
  // hg0 = relu(bn(h @ fcg_w0 + b0));  abf currently bf16(h)
  cvt_wT_bf16<<<nb((size_t)HID * HID), B, 0, stream>>>(fcg_w0, wt, HID);
  gemm(abf, wt, fcg_b + 0, t0, HID);
  bn(t0, bng_g + 0, bng_b + 0, hgA, abf);            // abf <- bf16(hg0)
  local_head(0, out + (size_t)1 * N_NODES * OUT_CH);

  // hg1 = relu(bn(concat(hg0,h) @ fcg_ws[0] + b1))
  cvt_concat_bf16<<<nb((size_t)N_NODES * IN_C), B, 0, stream>>>(hgA, h, abf, N_NODES * IN_C);
  cvt_wT_bf16<<<nb((size_t)HID * IN_C), B, 0, stream>>>(fcg_ws, wt, IN_C);
  gemm(abf, wt, fcg_b + HID, t0, IN_C);
  bn(t0, bng_g + HID, bng_b + HID, hgB, abf);        // abf <- bf16(hg1)
  local_head(1, out + (size_t)2 * N_NODES * OUT_CH);

  // hg2 = relu(bn(concat(hg1,h) @ fcg_ws[1] + b2))  (reuse hgA)
  cvt_concat_bf16<<<nb((size_t)N_NODES * IN_C), B, 0, stream>>>(hgB, h, abf, N_NODES * IN_C);
  cvt_wT_bf16<<<nb((size_t)HID * IN_C), B, 0, stream>>>(fcg_ws + (size_t)IN_C * HID, wt, IN_C);
  gemm(abf, wt, fcg_b + 2 * HID, t0, IN_C);
  bn(t0, bng_g + 2 * HID, bng_b + 2 * HID, hgA, abf); // abf <- bf16(hg2)
  local_head(2, out + (size_t)3 * N_NODES * OUT_CH);

  // og = log_softmax(hg2 @ outg_w + outg_b)
  head_logsoftmax<<<nb(N_NODES), B, 0, stream>>>(hgA, outg_w, outg_b, out);
}
